// WindowMSA_56916906606860
// MI455X (gfx1250) — matrix operand order repaired and checked by reading, not compiled, not run on hardware
//
#include <hip/hip_runtime.h>

typedef __attribute__((ext_vector_type(16))) __bf16 bf16x16;
typedef __attribute__((ext_vector_type(8)))  float  f32x8;

// ---------- helpers ----------

static __device__ __forceinline__ __bf16 f2bf(float f) {
    unsigned u = __builtin_bit_cast(unsigned, f);
    unsigned r = u + 0x7FFFu + ((u >> 16) & 1u);   // round-to-nearest-even
    unsigned short h = (unsigned short)(r >> 16);
    return __builtin_bit_cast(__bf16, h);
}

// A fragment (16x32, bf16), row-major source with given element stride.
// ISA 7.12.2: lane<16 holds M=lane, K={0..7,16..23}; lane>=16 holds M=lane-16, K={8..15,24..31}
static __device__ __forceinline__ bf16x16 load_a_frag(const __bf16* base, int stride, int lane) {
    const __bf16* p = base + (lane & 15) * stride;
    int half = lane >> 4;
    union { uint4 u[2]; bf16x16 v; } t;
    t.u[0] = *(const uint4*)(p + half * 8);        // K = half*8 .. +7
    t.u[1] = *(const uint4*)(p + 16 + half * 8);   // K = 16+half*8 .. +7
    return t.v;
}

// B fragment (32x16, bf16). Source stored "N rows x K contiguous".
// lane<16: N=lane, K={0..7,16..23}; lane>=16: N=lane-16, K={8..15,24..31}
static __device__ __forceinline__ bf16x16 load_b_frag(const __bf16* base, int stride, int lane) {
    const __bf16* p = base + (lane & 15) * stride + (lane >> 4) * 8;
    union { uint4 u[2]; bf16x16 v; } t;
    t.u[0] = *(const uint4*)(p);
    t.u[1] = *(const uint4*)(p + 16);
    return t.v;
}

static __device__ __forceinline__ f32x8 wmma_bf16(bf16x16 a, bf16x16 b, f32x8 c) {
    return __builtin_amdgcn_wmma_f32_16x16x32_bf16(false, a, false, b, (short)0, c, false, false);
}

// Async memory->LDS copy (16B per lane), tracked by ASYNCcnt (ISA 08 §4).
// LDS VGPR operand = byte offset (low 32 bits of flat shared address).
static __device__ __forceinline__ void async_copy_b128(const __bf16* gptr, __bf16* lptr) {
    unsigned ldsOff = (unsigned)(unsigned long long)lptr;
    asm volatile("global_load_async_to_lds_b128 %0, %1, off"
                 :: "v"(ldsOff), "v"(gptr)
                 : "memory");
}
static __device__ __forceinline__ void wait_asynccnt0() {
    asm volatile("s_wait_asynccnt 0x0" ::: "memory");
}

// ---------- kernel 1: qkv = x @ w1 + b1, scatter to q/k/vT (bf16) ----------
// x: [65536, 512] f32, w1: [512, 1536] f32
// q,k: [(b*16+h)*64+win][n][e]  (64x32 bf16 per slot), vT: [...][e][n] (32x64)

__global__ __launch_bounds__(256) void gemm_qkv_kernel(
    const float* __restrict__ x, const float* __restrict__ w1, const float* __restrict__ b1,
    __bf16* __restrict__ qb, __bf16* __restrict__ kb, __bf16* __restrict__ vtb) {
    __shared__ __bf16 As[128 * 32];   // [m][k]
    __shared__ __bf16 Bs[128 * 32];   // [n][k]  (K-transposed)
    const int tid  = threadIdx.x;
    const int lane = tid & 31, wave = tid >> 5;
    const int waveM = wave >> 1, waveN = wave & 1;
    const int mBase = blockIdx.y * 128;
    const int nBase = blockIdx.x * 128;
    const f32x8 fzero = {};

    f32x8 acc[2][4];
    for (int i = 0; i < 2; ++i)
        for (int j = 0; j < 4; ++j) acc[i][j] = fzero;

    for (int kk = 0; kk < 512; kk += 32) {
        __syncthreads();
        // X tile 128x32 f32 -> bf16 LDS
        #pragma unroll
        for (int i = 0; i < 4; ++i) {
            int idx = tid + i * 256;            // 1024 float4 chunks
            int row = idx >> 3, c4 = idx & 7;
            float4 v = *(const float4*)(x + (size_t)(mBase + row) * 512 + kk + c4 * 4);
            __bf16* dst = As + row * 32 + c4 * 4;
            dst[0] = f2bf(v.x); dst[1] = f2bf(v.y); dst[2] = f2bf(v.z); dst[3] = f2bf(v.w);
        }
        // W1 tile 32x128 f32 -> bf16 LDS, K-transposed
        #pragma unroll
        for (int i = 0; i < 4; ++i) {
            int idx = tid + i * 256;
            int kr = idx >> 5, c4 = idx & 31;
            float4 v = *(const float4*)(w1 + (size_t)(kk + kr) * 1536 + nBase + c4 * 4);
            int n0 = c4 * 4;
            Bs[(n0 + 0) * 32 + kr] = f2bf(v.x);
            Bs[(n0 + 1) * 32 + kr] = f2bf(v.y);
            Bs[(n0 + 2) * 32 + kr] = f2bf(v.z);
            Bs[(n0 + 3) * 32 + kr] = f2bf(v.w);
        }
        __syncthreads();
        if (kk + 32 < 512)
            __builtin_prefetch(x + (size_t)(mBase + (tid >> 3)) * 512 + kk + 32, 0, 1);

        bf16x16 aF[2];
        #pragma unroll
        for (int mt = 0; mt < 2; ++mt)
            aF[mt] = load_a_frag(As + (waveM * 32 + mt * 16) * 32, 32, lane);
        #pragma unroll
        for (int nt = 0; nt < 4; ++nt) {
            bf16x16 bF = load_b_frag(Bs + (waveN * 64 + nt * 16) * 32, 32, lane);
            #pragma unroll
            for (int mt = 0; mt < 2; ++mt)
                acc[mt][nt] = wmma_bf16(aF[mt], bF, acc[mt][nt]);
        }
    }

    // Epilogue: add bias, decompose col -> (s, e, h), row -> (b, window, token); scatter bf16
    const int half = lane >> 4, lc = lane & 15;
    for (int mt = 0; mt < 2; ++mt) {
        for (int nt = 0; nt < 4; ++nt) {
            int col = nBase + waveN * 64 + nt * 16 + lc;
            float bias = b1[col];
            int s = col % 3, ch = col / 3;
            int h = ch & 15, e = ch >> 4;                // c = e*16 + h ('(e H)' split)
            int row0 = mBase + waveM * 32 + mt * 16 + half * 8;
            #pragma unroll
            for (int r = 0; r < 8; ++r) {
                int R = row0 + r;
                int bb = R >> 12, l = R & 4095;
                int gy = l >> 6, gx = l & 63;
                int win = ((gy >> 3) << 3) + (gx >> 3);
                int n   = ((gy & 7) << 3) + (gx & 7);
                size_t base = (size_t)((bb * 16 + h) * 64 + win);
                __bf16 val = f2bf(acc[mt][nt][r] + bias);
                if (s == 0)      qb [base * 2048 + n * 32 + e] = val;
                else if (s == 1) kb [base * 2048 + n * 32 + e] = val;
                else             vtb[base * 2048 + e * 64 + n] = val;   // V transposed
            }
        }
    }
}

// ---------- kernel 2: windowed attention, one wave per (b, head, window) ----------

__global__ __launch_bounds__(128) void attn_kernel(
    const __bf16* __restrict__ qb, const __bf16* __restrict__ kb, const __bf16* __restrict__ vtb,
    const float* __restrict__ rel_bias, __bf16* __restrict__ attout) {
    __shared__ float  biasLds[225];                 // (2*8-1)^2 table
    __shared__ __bf16 Plds[4][64 * 64];             // per-wave probability tile
    const int tid = threadIdx.x;
    for (int i = tid; i < 225; i += 128) biasLds[i] = rel_bias[i];
    __syncthreads();

    const int lane = tid & 31, wave = tid >> 5;
    const int gw = blockIdx.x * 4 + wave;           // == (b*16 + h)*64 + win
    const __bf16* q  = qb  + (size_t)gw * 2048;
    const __bf16* k  = kb  + (size_t)gw * 2048;
    const __bf16* vt = vtb + (size_t)gw * 2048;
    const f32x8 fzero = {};

    bf16x16 aQ[4], bK[4];
    #pragma unroll
    for (int mt = 0; mt < 4; ++mt) aQ[mt] = load_a_frag(q + mt * 16 * 32, 32, lane);
    #pragma unroll
    for (int nt = 0; nt < 4; ++nt) bK[nt] = load_b_frag(k + nt * 16 * 32, 32, lane);

    f32x8 S[4][4];                                   // S = Q K^T, E=32 => single K-step
    #pragma unroll
    for (int mt = 0; mt < 4; ++mt)
        #pragma unroll
        for (int nt = 0; nt < 4; ++nt)
            S[mt][nt] = wmma_bf16(aQ[mt], bK[nt], fzero);

    const float scale = 0.1767766952966369f;         // 1/sqrt(32)
    const int half = lane >> 4, lc = lane & 15;
    __bf16* myP = Plds[wave];

    for (int mt = 0; mt < 4; ++mt) {
        #pragma unroll
        for (int r = 0; r < 8; ++r) {
            int m = mt * 16 + half * 8 + r;          // query token
            int mwh = m >> 3, mww = m & 7;
            float v[4];
            #pragma unroll
            for (int nt = 0; nt < 4; ++nt) {
                int n = nt * 16 + lc;                // key token
                int idx = (mwh - (n >> 3) + 7) * 15 + (mww - (n & 7) + 7);
                v[nt] = S[mt][nt][r] * scale + biasLds[idx];
            }
            // row-wise softmax: 4 local + 16-lane half-wave reduction (full 64 cols)
            float mx = fmaxf(fmaxf(v[0], v[1]), fmaxf(v[2], v[3]));
            for (int off = 8; off >= 1; off >>= 1) mx = fmaxf(mx, __shfl_xor(mx, off, 32));
            float sum = 0.f;
            #pragma unroll
            for (int nt = 0; nt < 4; ++nt) { v[nt] = __expf(v[nt] - mx); sum += v[nt]; }
            for (int off = 8; off >= 1; off >>= 1) sum += __shfl_xor(sum, off, 32);
            float inv = 1.0f / sum;
            #pragma unroll
            for (int nt = 0; nt < 4; ++nt) myP[m * 64 + nt * 16 + lc] = f2bf(v[nt] * inv);
        }
    }

    // O = P @ V : contraction over 64 tokens = 2 K-steps
    f32x8 O[4][2];
    for (int mt = 0; mt < 4; ++mt)
        for (int et = 0; et < 2; ++et) O[mt][et] = fzero;
    for (int kkk = 0; kkk < 64; kkk += 32) {
        bf16x16 aP[4];
        #pragma unroll
        for (int mt = 0; mt < 4; ++mt) aP[mt] = load_a_frag(myP + mt * 16 * 64 + kkk, 64, lane);
        #pragma unroll
        for (int et = 0; et < 2; ++et) {
            bf16x16 bV = load_b_frag(vt + et * 16 * 64 + kkk, 64, lane);
            #pragma unroll
            for (int mt = 0; mt < 4; ++mt) O[mt][et] = wmma_bf16(aP[mt], bV, O[mt][et]);
        }
    }

    // Store: out channel = h*32 + e  ('(H e)' merge), token -> global position
    const int bb = gw >> 10, h = (gw >> 6) & 15, win = gw & 63;
    const int wy = win >> 3, wx = win & 7;
    for (int mt = 0; mt < 4; ++mt) {
        int m0 = mt * 16 + half * 8;
        for (int et = 0; et < 2; ++et) {
            int e = et * 16 + lc;
            #pragma unroll
            for (int r = 0; r < 8; ++r) {
                int n = m0 + r;
                int gy = wy * 8 + (n >> 3), gx = wx * 8 + (n & 7);
                size_t l = (size_t)gy * 64 + gx;
                attout[((size_t)bb * 4096 + l) * 512 + h * 32 + e] = f2bf(O[mt][et][r]);
            }
        }
    }
}

// ---------- kernel 3: out = attout(bf16) @ w2 + b2 -> f32 ----------
// A tile is a raw bf16 copy -> use GLOBAL_LOAD_ASYNC_TO_LDS (ASYNCcnt path):
// the copy bypasses VGPRs and overlaps with the W2 f32->bf16 conversion stores.

__global__ __launch_bounds__(256) void gemm_out_kernel(
    const __bf16* __restrict__ a, const float* __restrict__ w2, const float* __restrict__ b2,
    float* __restrict__ out) {
    __shared__ __bf16 As[128 * 32];
    __shared__ __bf16 Bs[128 * 32];
    const int tid = threadIdx.x;
    const int lane = tid & 31, wave = tid >> 5;
    const int waveM = wave >> 1, waveN = wave & 1;
    const int mBase = blockIdx.y * 128;
    const int nBase = blockIdx.x * 128;
    const f32x8 fzero = {};

    f32x8 acc[2][4];
    for (int i = 0; i < 2; ++i)
        for (int j = 0; j < 4; ++j) acc[i][j] = fzero;

    for (int kk = 0; kk < 512; kk += 32) {
        __syncthreads();
        // A tile 128x32 bf16: async memory->LDS, 16B/lane, 2 chunks per thread
        #pragma unroll
        for (int i = 0; i < 2; ++i) {
            int idx = tid + i * 256;                 // 512 uint4 chunks
            int row = idx >> 2, c8 = idx & 3;
            async_copy_b128(a + (size_t)(mBase + row) * 512 + kk + c8 * 8,
                            As + row * 32 + c8 * 8);
        }
        // W2 tile f32 -> bf16, K-transposed (overlaps with async copies above)
        #pragma unroll
        for (int i = 0; i < 4; ++i) {
            int idx = tid + i * 256;
            int kr = idx >> 5, c4 = idx & 31;
            float4 v = *(const float4*)(w2 + (size_t)(kk + kr) * 512 + nBase + c4 * 4);
            int n0 = c4 * 4;
            Bs[(n0 + 0) * 32 + kr] = f2bf(v.x);
            Bs[(n0 + 1) * 32 + kr] = f2bf(v.y);
            Bs[(n0 + 2) * 32 + kr] = f2bf(v.z);
            Bs[(n0 + 3) * 32 + kr] = f2bf(v.w);
        }
        wait_asynccnt0();                            // async tiles landed in LDS
        __syncthreads();
        if (kk + 32 < 512)
            __builtin_prefetch(a + (size_t)(mBase + (tid >> 2)) * 512 + kk + 32, 0, 1);

        bf16x16 aF[2];
        #pragma unroll
        for (int mt = 0; mt < 2; ++mt)
            aF[mt] = load_a_frag(As + (waveM * 32 + mt * 16) * 32, 32, lane);
        #pragma unroll
        for (int nt = 0; nt < 4; ++nt) {
            bf16x16 bF = load_b_frag(Bs + (waveN * 64 + nt * 16) * 32, 32, lane);
            #pragma unroll
            for (int mt = 0; mt < 2; ++mt)
                acc[mt][nt] = wmma_bf16(aF[mt], bF, acc[mt][nt]);
        }
    }

    const int half = lane >> 4, lc = lane & 15;
    for (int mt = 0; mt < 2; ++mt) {
        int row0 = mBase + waveM * 32 + mt * 16 + half * 8;
        for (int nt = 0; nt < 4; ++nt) {
            int col = nBase + waveN * 64 + nt * 16 + lc;
            float bias = b2[col];
            #pragma unroll
            for (int r = 0; r < 8; ++r)
                out[(size_t)(row0 + r) * 512 + col] = acc[mt][nt][r] + bias;
        }
    }
}

// ---------- launch ----------

extern "C" void kernel_launch(void* const* d_in, const int* in_sizes, int n_in,
                              void* d_out, int out_size, void* d_ws, size_t ws_size,
                              hipStream_t stream) {
    const float* x  = (const float*)d_in[0];
    const float* w1 = (const float*)d_in[1];
    const float* b1 = (const float*)d_in[2];
    const float* w2 = (const float*)d_in[3];
    const float* b2 = (const float*)d_in[4];
    const float* rb = (const float*)d_in[5];
    float* out = (float*)d_out;

    char* ws = (char*)d_ws;
    const size_t SEG = (size_t)16 * 16 * 64 * 64 * 32 * 2;  // 64 MB per buffer
    __bf16* qb  = (__bf16*)(ws);
    __bf16* kb  = (__bf16*)(ws + SEG);
    __bf16* vtb = (__bf16*)(ws + 2 * SEG);
    __bf16* ao  = (__bf16*)(ws + 3 * SEG);

    gemm_qkv_kernel<<<dim3(12, 512), 256, 0, stream>>>(x, w1, b1, qb, kb, vtb);
    attn_kernel<<<dim3(4096), 128, 0, stream>>>(qb, kb, vtb, rb, ao);
    gemm_out_kernel<<<dim3(4, 512), 256, 0, stream>>>(ao, w2, b2, out);
}
